// progressive_social_stgcnn_73778948210701
// MI455X (gfx1250) — matrix operand order repaired and checked by reading, not compiled, not run on hardware
//
#include <hip/hip_runtime.h>
#include <hip/hip_bf16.h>

// ---------------------------------------------------------------------------
// progressive social-stgcnn forward for MI455X (gfx1250), bf16 WMMA datapath
//  - einsum as 8 GEMMs (1280x1024)x(1024x1024): WMMA, b128-vector A/B loads,
//    4-way N register blocking, explicit 2-deep software pipeline
//  - TXP 3x3 convs (+fused laterals) as implicit GEMM: im2col staged in LDS
//    once per block; K-loop fully unrolled via KPAD template
// ---------------------------------------------------------------------------

typedef __attribute__((ext_vector_type(16))) __bf16 v16bf;
typedef __attribute__((ext_vector_type(8)))  float  v8f;

union BFrag { v16bf v; unsigned short s[16]; uint4 q[2]; };

#define DEV static __device__ __forceinline__

DEV unsigned short f2bf(float f) {               // round-to-nearest-even f32->bf16
  unsigned u = __float_as_uint(f);
  u += 0x7FFFu + ((u >> 16) & 1u);
  return (unsigned short)(u >> 16);
}
DEV float bf2f(unsigned short h) { return __uint_as_float(((unsigned)h) << 16); }
DEV float prelu_f(float x, float a) { return x >= 0.f ? x : a * x; }
DEV float sel5(const float* a, int c) {
  return c == 0 ? a[0] : c == 1 ? a[1] : c == 2 ? a[2] : c == 3 ? a[3] : a[4];
}
DEV void load_af(BFrag& f, const unsigned short* base) {   // A frag: 16 halves
  const uint4* p = reinterpret_cast<const uint4*>(base);
  f.q[0] = p[0]; f.q[1] = p[1];
}
DEV void load_bf(BFrag& f, const unsigned short* base) {   // B frag: 16 halves
  const uint4* p = reinterpret_cast<const uint4*>(base);
  f.q[0] = p[0]; f.q[1] = p[1];
}
#define WMMA_BF16(ACC, A, B) \
  ACC = __builtin_amdgcn_wmma_f32_16x16x32_bf16(false, (A).v, false, (B).v, (short)0, ACC, false, false)

#define NB   256      // batch
#define VV_  1024     // graph nodes
#define TT_  8        // observed seq
#define INF_ 2
#define OUTF 5
#define PREDC 12
#define ME   (NB*OUTF)          // 1280 einsum GEMM rows
#define EPSB 1e-5f

// ---------------------------------------------------------------------------
__global__ void k_zero_u16(unsigned short* d, int n) {
  int i = blockIdx.x * blockDim.x + threadIdx.x;
  if (i < n) d[i] = 0;
}

// a (T,V,V) f32 -> aT (T,V,V) bf16 with v<->w transpose, LDS-tiled
__global__ void k_cvt_transpose(const float* __restrict__ a, unsigned short* __restrict__ aT) {
  __shared__ unsigned short tile[32][33];
  int bid  = blockIdx.x;                       // 8 * 32 * 32 blocks
  int wblk = bid & 31;
  int vblk = (bid >> 5) & 31;
  int t    = bid >> 10;
  int tx = threadIdx.x & 31, ty = threadIdx.x >> 5;   // (32,8)
#pragma unroll
  for (int rr = 0; rr < 32; rr += 8) {
    int vvv = vblk * 32 + ty + rr;
    int www = wblk * 32 + tx;
    tile[ty + rr][tx] = f2bf(a[((size_t)t * VV_ + vvv) * VV_ + www]);
  }
  __syncthreads();
#pragma unroll
  for (int rr = 0; rr < 32; rr += 8) {
    int www = wblk * 32 + ty + rr;
    int vvv = vblk * 32 + tx;
    aT[((size_t)t * VV_ + www) * VV_ + vvv] = tile[tx][ty + rr];
  }
}

// gcn 1x1 conv (bf16, einsum-A layout) + residual 1x1 conv + resbn (bf16)
__global__ void k_gcn_res(const float* __restrict__ v,
                          const float* gw, const float* gb,
                          const float* rw, const float* rb,
                          const float* rg, const float* rbb, const float* rm, const float* rv,
                          unsigned short* __restrict__ xg, unsigned short* __restrict__ res) {
  int i  = blockIdx.x * blockDim.x + threadIdx.x;   // over N*T*V (exact)
  int vv = i % VV_;
  int t  = (i / VV_) % TT_;
  int n  = i / (VV_ * TT_);
  float x0 = v[((size_t)(n * INF_ + 0) * TT_ + t) * VV_ + vv];
  float x1 = v[((size_t)(n * INF_ + 1) * TT_ + t) * VV_ + vv];
#pragma unroll
  for (int c = 0; c < OUTF; ++c) {
    float g = gw[c * 2] * x0 + gw[c * 2 + 1] * x1 + gb[c];
    xg[((size_t)t * ME + (n * OUTF + c)) * VV_ + vv] = f2bf(g);
    float r = rw[c * 2] * x0 + rw[c * 2 + 1] * x1 + rb[c];
    float sc = rsqrtf(rv[c] + EPSB);
    r = (r - rm[c]) * sc * rg[c] + rbb[c];
    res[((size_t)(n * OUTF + c) * TT_ + t) * VV_ + vv] = f2bf(r);
  }
}

// einsum 'nctv,tvw->nctw' : per wave one 16-row M tile x four 16-col N tiles.
// A from xg [t][row][v], B from aT [t][w][v] (both K-contiguous -> b128 loads).
// 2-deep software pipeline so prefetch loads overlap the WMMA block.
__global__ __launch_bounds__(256) void k_einsum_wmma(
    const unsigned short* __restrict__ xg, const unsigned short* __restrict__ aT,
    const float* g1, const float* b1, const float* m1, const float* vr1,
    const float* pa, unsigned short* __restrict__ y1) {
  int wave = threadIdx.x >> 5, lane = threadIdx.x & 31;
  int tid  = blockIdx.x * 8 + wave;                 // 10240 = 80 mt * 16 ng * 8 t
  int ng = tid & 15;
  int mt = (tid >> 4) % 80;
  int t  = tid / (16 * 80);
  int hb  = lane >> 4;
  int l16 = lane & 15;
  int klo   = hb ? 16 : 0;                          // A frag K offset (contiguous k = 16h + s)
  int khalf = hb ? 16 : 0;                          // B frag K offset (contiguous k = 16h + s)
  const unsigned short* arow = xg + ((size_t)t * ME + (mt * 16 + l16)) * VV_ + klo;
  const unsigned short* brow[4];
#pragma unroll
  for (int j = 0; j < 4; ++j)
    brow[j] = aT + ((size_t)t * VV_ + (ng * 64 + j * 16 + l16)) * VV_ + khalf;

  v8f acc0 = {}, acc1 = {}, acc2 = {}, acc3 = {};
  BFrag aP, aQ, bP0, bP1, bP2, bP3, bQ0, bQ1, bQ2, bQ3;

  load_af(aP, arow);
  load_bf(bP0, brow[0]); load_bf(bP1, brow[1]);
  load_bf(bP2, brow[2]); load_bf(bP3, brow[3]);
#pragma unroll 1
  for (int kb = 0; kb < 30; kb += 2) {
    int o1 = (kb + 1) * 32, o2 = (kb + 2) * 32;
    load_af(aQ, arow + o1);
    load_bf(bQ0, brow[0] + o1); load_bf(bQ1, brow[1] + o1);
    load_bf(bQ2, brow[2] + o1); load_bf(bQ3, brow[3] + o1);
    WMMA_BF16(acc0, aP, bP0); WMMA_BF16(acc1, aP, bP1);
    WMMA_BF16(acc2, aP, bP2); WMMA_BF16(acc3, aP, bP3);
    load_af(aP, arow + o2);
    load_bf(bP0, brow[0] + o2); load_bf(bP1, brow[1] + o2);
    load_bf(bP2, brow[2] + o2); load_bf(bP3, brow[3] + o2);
    WMMA_BF16(acc0, aQ, bQ0); WMMA_BF16(acc1, aQ, bQ1);
    WMMA_BF16(acc2, aQ, bQ2); WMMA_BF16(acc3, aQ, bQ3);
  }
  {                                                  // peeled tail: kb = 30, 31
    int o1 = 31 * 32;
    load_af(aQ, arow + o1);
    load_bf(bQ0, brow[0] + o1); load_bf(bQ1, brow[1] + o1);
    load_bf(bQ2, brow[2] + o1); load_bf(bQ3, brow[3] + o1);
    WMMA_BF16(acc0, aP, bP0); WMMA_BF16(acc1, aP, bP1);
    WMMA_BF16(acc2, aP, bP2); WMMA_BF16(acc3, aP, bP3);
    WMMA_BF16(acc0, aQ, bQ0); WMMA_BF16(acc1, aQ, bQ1);
    WMMA_BF16(acc2, aQ, bQ2); WMMA_BF16(acc3, aQ, bQ3);
  }

  // bn1 scale/shift in registers; channel of row gm: 16 == 1 (mod 5)
  float sc5[5], sh5[5];
#pragma unroll
  for (int c = 0; c < OUTF; ++c) {
    float s = rsqrtf(vr1[c] + EPSB) * g1[c];
    sc5[c] = s;
    sh5[c] = b1[c] - m1[c] * s;
  }
  float pav = pa[0];
  int cb = mt % 5 + 3 * hb; cb = cb >= 5 ? cb - 5 : cb;
  v8f accs[4] = {acc0, acc1, acc2, acc3};
#pragma unroll
  for (int j = 0; j < 4; ++j) {
    int gn = ng * 64 + j * 16 + l16;
#pragma unroll
    for (int r = 0; r < 8; ++r) {
      int gm = mt * 16 + r + hb * 8;
      int c = cb + r; c = c >= 5 ? c - 5 : c; c = c >= 5 ? c - 5 : c;
      float y = accs[j][r] * sel5(sc5, c) + sel5(sh5, c);
      y = prelu_f(y, pav);
      y1[((size_t)gm * TT_ + t) * VV_ + gn] = f2bf(y);
    }
  }
}

// tcn (3,1) conv pad(1,0) + bn2 + residual add + PReLU -> bf16 TXP level-1 input
// One thread per (n,t,vv): loads the 5x3 patch once, emits all 5 channels.
__global__ void k_tcn(const unsigned short* __restrict__ y1,
                      const unsigned short* __restrict__ res,
                      const float* tw, const float* tb,
                      const float* g2, const float* b2, const float* m2, const float* v2,
                      const float* pp, unsigned short* __restrict__ out) {
  int i  = blockIdx.x * blockDim.x + threadIdx.x;   // N*T*V (exact)
  int vv = i % VV_;
  int t  = (i / VV_) % TT_;
  int n  = i / (VV_ * TT_);
  float xin[OUTF][3];
#pragma unroll
  for (int ci = 0; ci < OUTF; ++ci)
#pragma unroll
    for (int dt = 0; dt < 3; ++dt) {
      int tt = t + dt - 1;
      xin[ci][dt] = (tt >= 0 && tt < TT_)
                        ? bf2f(y1[((size_t)(n * OUTF + ci) * TT_ + tt) * VV_ + vv])
                        : 0.f;
    }
  float pav = pp[0];
#pragma unroll
  for (int c = 0; c < OUTF; ++c) {
    float acc = tb[c];
#pragma unroll
    for (int ci = 0; ci < OUTF; ++ci)
#pragma unroll
      for (int dt = 0; dt < 3; ++dt)
        acc += xin[ci][dt] * tw[(c * OUTF + ci) * 3 + dt];
    float sc = rsqrtf(v2[c] + EPSB);
    float y  = (acc - m2[c]) * sc * g2[c] + b2[c];
    size_t off = ((size_t)(n * OUTF + c) * TT_ + t) * VV_ + vv;
    y += bf2f(res[off]);
    out[off] = f2bf(prelu_f(y, pav));
  }
}

// pack conv weights (12,cin,3,3) f32 -> bf16 lane-contiguous B-fragment layout:
// dst[((kb*2 + h)*16 + co)*16 + i]  where k = kb*32 + h*16 + i
__global__ void k_pack_w(const float* __restrict__ w, unsigned short* __restrict__ dst,
                         int cin, int kbase) {
  int i = blockIdx.x * blockDim.x + threadIdx.x;
  int tot = cin * 9 * PREDC;
  if (i >= tot) return;
  int co = i % PREDC;
  int kr = i / PREDC;                               // ci_*9 + kh*3 + kw
  int k  = kbase + kr;
  int kb = k >> 5, rem = k & 31, h = rem >> 4, ii = rem & 15;
  dst[(size_t)(((kb * 2 + h) * 16 + co) * 16) + ii] = f2bf(w[co * cin * 9 + kr]);
}

// TXP 3x3 conv (+ fused lateral via concatenated K) as implicit-GEMM WMMA.
// Block = 128 consecutive vv rows (single (n,d)); im2col tile staged in LDS
// [row][KPAD] with zero halo; fully unrolled K loop (KPAD template).
// mode: 0 = prelu, 1 = prelu + residual, 2 = final f32 store (reshape = flat copy)
template <int KPAD>
__global__ __launch_bounds__(256) void k_txp_wmma(
    const unsigned short* __restrict__ in1, int cin1,
    const unsigned short* __restrict__ in0, int cin0,
    const unsigned short* __restrict__ wp, int Kblk,
    const float* __restrict__ bias1, const float* __restrict__ bias0,
    int mode, unsigned short* __restrict__ outb, float* __restrict__ outf) {
  extern __shared__ unsigned short lds_a[];         // 128 * KPAD halves
  int wave = threadIdx.x >> 5, lane = threadIdx.x & 31;
  int hb = lane >> 4, l16 = lane & 15;
  int row0  = blockIdx.x * 128;                     // 10240 blocks (exact)
  int n     = row0 / (5 * VV_);
  int d     = (row0 / VV_) % 5;
  int vvblk = row0 % VV_;                           // multiple of 128

  // ---- stage im2col tile: decode each k once, coalesced reads, zero halo ----
  for (int idx = threadIdx.x; idx < KPAD * 128; idx += 256) {
    int k   = idx >> 7;
    int lvv = idx & 127;
    unsigned short hv = 0;
    const unsigned short* src = nullptr;
    int rel = k, cs = cin1;
    if (k < Kblk) { src = in1; }
    else if (in0) { rel = k - Kblk; if (rel < cin0 * 9) { src = in0; cs = cin0; } }
    if (src) {
      int ci_ = rel / 9, t9 = rel % 9, kh = t9 / 3, kw = t9 % 3;
      int hh = d + kh - 1, ww = vvblk + lvv + kw - 1;
      if (hh >= 0 && hh < 5 && (unsigned)ww < (unsigned)VV_)
        hv = src[(size_t)n * cs * 5 * VV_ + (size_t)(ci_ * 5 + hh) * VV_ + ww];
    }
    lds_a[lvv * KPAD + k] = hv;
  }
  __syncthreads();

  // ---- GEMM: 1 M-tile per wave, cols = 12 output channels (padded to 16) ----
  int klo = hb ? 16 : 0;
  const unsigned short* arow = lds_a + (wave * 16 + l16) * KPAD + klo;
  const unsigned short* bcol = wp + (size_t)(hb * 16 + l16) * 16;
  v8f acc = {};
#pragma unroll
  for (int kb = 0; kb < KPAD / 32; ++kb) {
    BFrag af, bw;
    load_af(af, arow + kb * 32);
    load_bf(bw, bcol + kb * 512);                   // (kb*2)*16*16 halves
    WMMA_BF16(acc, af, bw);
  }

  int co = l16;
  if (co < PREDC) {
    float bias = bias1[co] + (bias0 ? bias0[co] : 0.f);
#pragma unroll
    for (int r = 0; r < 8; ++r) {
      int vv = vvblk + wave * 16 + r + hb * 8;
      float y = acc[r] + bias;
      size_t off = (size_t)n * PREDC * 5 * VV_ + (size_t)(co * 5 + d) * VV_ + vv;
      if (mode == 2) {
        outf[off] = y;
      } else {
        y = prelu_f(y, 0.25f);
        if (mode == 1)
          y += bf2f(in1[(size_t)n * cin1 * 5 * VV_ + (size_t)(co * 5 + d) * VV_ + vv]);
        outb[off] = f2bf(y);
      }
    }
  }
}

// ---------------------------------------------------------------------------
extern "C" void kernel_launch(void* const* d_in, const int* in_sizes, int n_in,
                              void* d_out, int out_size, void* d_ws, size_t ws_size,
                              hipStream_t stream) {
  // Leaf order (depth-first, dict insertion order as written, None skipped):
  // 0:v 1:a ; per column (col0 @2, col1 @32):
  //  +0 gcn_w +1 gcn_b +2..5 bn1{g,b,m,v} +6 tcn_prelu +7 tcn_w +8 tcn_b
  //  +9..12 bn2{g,b,m,v} +13 res_w +14 res_b +15..18 resbn{g,b,m,v} +19 prelu
  //  +20..29 blocks(w,b)x5 ; col1 only: +30..39 lat(w,b)x5
  const float* vin = (const float*)d_in[0];
  const float* a   = (const float*)d_in[1];
  auto F = [&](int i) { return (const float*)d_in[i]; };
  const int CB[2] = {2, 32};

  char* ws = (char*)d_ws;
  size_t cur = 0;
  auto carve = [&](size_t bytes) -> char* {
    char* p = ws + cur;
    cur = (cur + bytes + 255) & ~(size_t)255;
    return p;
  };
  unsigned short* aT  = (unsigned short*)carve((size_t)TT_ * VV_ * VV_ * 2);
  unsigned short* xg  = (unsigned short*)carve((size_t)TT_ * ME * VV_ * 2);
  unsigned short* res = (unsigned short*)carve((size_t)NB * OUTF * TT_ * VV_ * 2);
  unsigned short* y1  = (unsigned short*)carve((size_t)NB * OUTF * TT_ * VV_ * 2);
  unsigned short* buf[2][2];
  for (int ci = 0; ci < 2; ++ci)
    for (int p = 0; p < 2; ++p)
      buf[ci][p] = (unsigned short*)carve((size_t)NB * PREDC * 5 * VV_ * 2);

  const int KpadT[2][5] = {{96, 128, 128, 128, 128}, {160, 224, 224, 224, 224}};
  size_t wo[2][5], wtot = 0;
  for (int ci = 0; ci < 2; ++ci)
    for (int l = 0; l < 5; ++l) { wo[ci][l] = wtot; wtot += (size_t)KpadT[ci][l] * 16; }
  unsigned short* wpak = (unsigned short*)carve(wtot * 2);

  // 1) adjacency f32 -> bf16, transposed to [t][w][v] (L2-resident, 16 MB)
  k_cvt_transpose<<<TT_ * 32 * 32, 256, 0, stream>>>(a, aT);

  // 2) pack all TXP weights into lane-contiguous fragment layout
  k_zero_u16<<<(int)((wtot + 255) / 256), 256, 0, stream>>>(wpak, (int)wtot);
  for (int ci = 0; ci < 2; ++ci)
    for (int l = 0; l < 5; ++l) {
      int cin = (l == 0) ? TT_ : PREDC;
      int thr = cin * 9 * PREDC;
      k_pack_w<<<(thr + 255) / 256, 256, 0, stream>>>(F(CB[ci] + 20 + 2 * l),
                                                      wpak + wo[ci][l], cin, 0);
      if (ci == 1)
        k_pack_w<<<(thr + 255) / 256, 256, 0, stream>>>(F(CB[ci] + 30 + 2 * l),
                                                        wpak + wo[ci][l], cin, cin * 9);
    }

  // 3) st-gcn per column: gcn/res convs -> WMMA einsum (+bn1+prelu) -> tcn
  for (int ci = 0; ci < 2; ++ci) {
    int b = CB[ci];
    k_gcn_res<<<(NB * TT_ * VV_) / 256, 256, 0, stream>>>(
        vin, F(b + 0), F(b + 1), F(b + 13), F(b + 14),
        F(b + 15), F(b + 16), F(b + 17), F(b + 18), xg, res);
    k_einsum_wmma<<<(80 * 16 * 8) / 8, 256, 0, stream>>>(
        xg, aT, F(b + 2), F(b + 3), F(b + 4), F(b + 5), F(b + 6), y1);
    k_tcn<<<(NB * TT_ * VV_) / 256, 256, 0, stream>>>(
        y1, res, F(b + 7), F(b + 8), F(b + 9), F(b + 10), F(b + 11), F(b + 12),
        F(b + 19), buf[ci][0]);
  }

  // 4) TXP-CNN: 5 levels, ping-pong buffers, lateral fused into K for col1
  for (int l = 1; l <= 5; ++l) {
    int p = (l - 1) & 1;
    int cin = (l == 1) ? TT_ : PREDC;
    for (int ci = 0; ci < 2; ++ci) {
      if (ci == 0 && l == 5) continue;          // col0 level-5 output unused
      int b = CB[ci];
      const float* b1 = F(b + 21 + 2 * (l - 1));
      const float* b0 = (ci == 1) ? F(b + 31 + 2 * (l - 1)) : nullptr;
      const unsigned short* i0 = (ci == 1) ? buf[0][p] : nullptr;
      int mode = (l == 1) ? 0 : (l < 5 ? 1 : 2);
      int kpad = KpadT[ci][l - 1];
      size_t shbytes = (size_t)kpad * 128 * 2;
      int grid = (NB * 5 * VV_) / 128;
      unsigned short* ob = buf[ci][1 - p];
      switch (kpad) {
        case 96:
          k_txp_wmma<96><<<grid, 256, shbytes, stream>>>(
              buf[ci][p], cin, i0, cin, wpak + wo[ci][l - 1], cin * 9,
              b1, b0, mode, ob, (float*)d_out);
          break;
        case 128:
          k_txp_wmma<128><<<grid, 256, shbytes, stream>>>(
              buf[ci][p], cin, i0, cin, wpak + wo[ci][l - 1], cin * 9,
              b1, b0, mode, ob, (float*)d_out);
          break;
        case 160:
          k_txp_wmma<160><<<grid, 256, shbytes, stream>>>(
              buf[ci][p], cin, i0, cin, wpak + wo[ci][l - 1], cin * 9,
              b1, b0, mode, ob, (float*)d_out);
          break;
        default:
          k_txp_wmma<224><<<grid, 256, shbytes, stream>>>(
              buf[ci][p], cin, i0, cin, wpak + wo[ci][l - 1], cin * 9,
              b1, b0, mode, ob, (float*)d_out);
          break;
      }
    }
  }
}
